// Seq2Seq_33913061769817
// MI455X (gfx1250) — matrix operand order repaired and checked by reading, not compiled, not run on hardware
//
#include <hip/hip_runtime.h>
#include <hip/hip_bf16.h>
#include <math.h>

typedef __attribute__((ext_vector_type(16))) __bf16 v16bf;
typedef __attribute__((ext_vector_type(8)))  __bf16 v8bf;
typedef __attribute__((ext_vector_type(8)))  float  v8f;

namespace {
constexpr int    Bn = 32, Sn = 128, Tn = 64, Vn = 32000, En = 512, Hn = 512;
constexpr int    Kc = En + Hn;            // 1024 (combined [x|h] K dim)
constexpr int    Gc = 4 * Hn;             // 2048 (gates)
constexpr size_t WC_ELEMS = (size_t)Gc * Kc;           // per-cell combined weight elems
constexpr size_t NWC      = 4 * WC_ELEMS;              // 8,388,608
constexpr size_t NCLS     = (size_t)Vn * Kc;           // 32,768,000
constexpr size_t NBIAS    = 4 * (size_t)Gc;            // 8,192
constexpr size_t NSTATE   = 2ull * Bn * Hn * 2;        // h(2 cells)+c(2 cells) floats
constexpr size_t NOUT0    = (size_t)Bn * Vn;           // zero slice t=0

// workspace byte offsets (all 256B aligned)
constexpr size_t OFF_WC   = 0;                          // 4x [2048,1024] bf16
constexpr size_t OFF_WCLS = OFF_WC   + NWC * 2;         // [32000,1024] bf16
constexpr size_t OFF_BIAS = OFF_WCLS + NCLS * 2;        // 4x [2048] f32
constexpr size_t OFF_H    = OFF_BIAS + NBIAS * 4;       // 2x [32,512] f32
constexpr size_t OFF_C    = OFF_H + 2ull * Bn * Hn * 4; // 2x [32,512] f32
constexpr size_t OFF_X    = OFF_C + 2ull * Bn * Hn * 4; // 2x [32,1024] bf16
constexpr size_t OFF_HCAT = OFF_X + 2ull * Bn * Kc * 2; // [32,1024] bf16
constexpr size_t OFF_TOK  = OFF_HCAT + (size_t)Bn * Kc * 2; // [32] int
} // namespace

union ABfrag { v16bf v; v8bf h[2]; };

__device__ __forceinline__ float sigmoid_f(float x) { return 1.0f / (1.0f + expf(-x)); }

// ---------------------------------------------------------------------------
// Prep: bf16 weight conversion/packing, fused biases, zero states & t=0 output
// ---------------------------------------------------------------------------
__global__ void __launch_bounds__(256) prep_kernel(
    const float* wih0, const float* whh0, const float* wih1, const float* whh1,
    const float* wih2, const float* whh2, const float* wih3, const float* whh3,
    const float* bih0, const float* bhh0, const float* bih1, const float* bhh1,
    const float* bih2, const float* bhh2, const float* bih3, const float* bhh3,
    const float* clsw,
    __bf16* Wc, __bf16* Wcls, float* bias, float* hc_states, float* out)
{
  const float* wih[4] = {wih0, wih1, wih2, wih3};
  const float* whh[4] = {whh0, whh1, whh2, whh3};
  const float* bih[4] = {bih0, bih1, bih2, bih3};
  const float* bhh[4] = {bhh0, bhh1, bhh2, bhh3};
  const size_t total = NWC + NCLS + NBIAS + NSTATE + NOUT0;
  const size_t stride = (size_t)gridDim.x * blockDim.x;
  for (size_t i = (size_t)blockIdx.x * blockDim.x + threadIdx.x; i < total; i += stride) {
    size_t idx = i;
    if (idx < NWC) {
      // combined weight: Wc[cell][n][k], k<512 -> wih[n][k], else whh[n][k-512]
      int c   = (int)(idx / WC_ELEMS);
      size_t r = idx % WC_ELEMS;
      int n = (int)(r / Kc), k = (int)(r % Kc);
      float v = (k < En) ? wih[c][(size_t)n * En + k]
                         : whh[c][(size_t)n * Hn + (k - En)];
      Wc[idx] = (__bf16)v;
    } else if ((idx -= NWC) < NCLS) {
      Wcls[idx] = (__bf16)clsw[idx];            // already [V, 2H] output-major
    } else if ((idx -= NCLS) < NBIAS) {
      int c = (int)(idx / Gc), n = (int)(idx % Gc);
      bias[idx] = bih[c][n] + bhh[c][n];
    } else if ((idx -= NBIAS) < NSTATE) {
      hc_states[idx] = 0.0f;                    // h then c, contiguous
    } else {
      idx -= NSTATE;
      size_t b = idx / Vn, v = idx % Vn;
      out[(b * Tn + 0) * (size_t)Vn + v] = 0.0f; // zeros at t=0
    }
  }
}

// ---------------------------------------------------------------------------
// Build Xcat = bf16 [x=emb[token] | h] for two cells (gridDim.y selects cell)
// ---------------------------------------------------------------------------
__global__ void __launch_bounds__(256) build_xcat2_kernel(
    const float* __restrict__ emb,
    const int* tokA, int strideA, int offA, const float* hA, __bf16* XA,
    const int* tokB, int strideB, int offB, const float* hB, __bf16* XB)
{
  const int cell = blockIdx.y;
  const int* tok  = cell ? tokB : tokA;
  const int  ts   = cell ? strideB : strideA;
  const int  toff = cell ? offB : offA;
  const float* h  = cell ? hB : hA;
  __bf16* X       = cell ? XB : XA;

  int idx = blockIdx.x * blockDim.x + threadIdx.x;   // 0 .. 32*1024-1
  int b = idx >> 10, k = idx & (Kc - 1);
  float v;
  if (k < En) {
    int t = tok[(size_t)b * ts + toff];
    v = emb[(size_t)t * En + k];
  } else {
    v = h[(size_t)b * Hn + (k - En)];
  }
  X[idx] = (__bf16)v;
}

// ---------------------------------------------------------------------------
// Fused LSTM cell step (WMMA bf16 -> f32). Each wave owns one 16-wide hidden
// slice and computes ALL FOUR gate tiles (shared A fragment), then performs
// the sigmoid/tanh c/h update in-register. gridDim.y selects one of 2 cells.
// ---------------------------------------------------------------------------
__global__ void __launch_bounds__(256) lstm_cell2_kernel(
    const __bf16* X0, const __bf16* W0, const float* b0, float* h0, float* c0,
    __bf16* hcat0, int hoff0,
    const __bf16* X1, const __bf16* W1, const float* b1, float* h1, float* c1,
    __bf16* hcat1, int hoff1)
{
  const int cell = blockIdx.y;
  const __bf16* X    = cell ? X1 : X0;
  const __bf16* W    = cell ? W1 : W0;
  const float*  bias = cell ? b1 : b0;
  float* h           = cell ? h1 : h0;
  float* c           = cell ? c1 : c0;
  __bf16* hcat       = cell ? hcat1 : hcat0;
  const int hoff     = cell ? hoff1 : hoff0;

  const int wave  = blockIdx.x * 8 + (threadIdx.x >> 5);  // 64 waves
  const int lane  = threadIdx.x & 31;
  const int mtile = wave & 1;        // batch tile (0..1 for B=32)
  const int ntile = wave >> 1;       // hidden tile (0..31 for H=512)

  // A fragment (16x32 bf16): lane = M row, kbase = 16*(lane/16), contiguous k order
  const int mrow  = mtile * 16 + (lane & 15);
  const int kbase = (lane >> 4) * 16;
  // B fragment (32x16 bf16): lane = N column, koffB = 16*(lane/16), same k order as A
  const int nb    = lane & 15;
  const int koffB = (lane >> 4) * 16;

  const __bf16* arow = X + (size_t)mrow * Kc;
  const __bf16* bi = W + ((size_t)(0 * Hn + ntile * 16 + nb)) * Kc;
  const __bf16* bf = W + ((size_t)(1 * Hn + ntile * 16 + nb)) * Kc;
  const __bf16* bg = W + ((size_t)(2 * Hn + ntile * 16 + nb)) * Kc;
  const __bf16* bo = W + ((size_t)(3 * Hn + ntile * 16 + nb)) * Kc;

  v8f ai = {}, af = {}, ag = {}, ao = {};
  for (int k = 0; k < Kc; k += 32) {
    ABfrag a, wfi, wff, wfg, wfo;
    a.h[0]   = *(const v8bf*)(arow + k + kbase);
    a.h[1]   = *(const v8bf*)(arow + k + kbase + 8);
    wfi.h[0] = *(const v8bf*)(bi + k + koffB);
    wfi.h[1] = *(const v8bf*)(bi + k + koffB + 8);
    wff.h[0] = *(const v8bf*)(bf + k + koffB);
    wff.h[1] = *(const v8bf*)(bf + k + koffB + 8);
    wfg.h[0] = *(const v8bf*)(bg + k + koffB);
    wfg.h[1] = *(const v8bf*)(bg + k + koffB + 8);
    wfo.h[0] = *(const v8bf*)(bo + k + koffB);
    wfo.h[1] = *(const v8bf*)(bo + k + koffB + 8);
    ai = __builtin_amdgcn_wmma_f32_16x16x32_bf16(false, a.v, false, wfi.v, (short)0, ai, false, false);
    af = __builtin_amdgcn_wmma_f32_16x16x32_bf16(false, a.v, false, wff.v, (short)0, af, false, false);
    ag = __builtin_amdgcn_wmma_f32_16x16x32_bf16(false, a.v, false, wfg.v, (short)0, ag, false, false);
    ao = __builtin_amdgcn_wmma_f32_16x16x32_bf16(false, a.v, false, wfo.v, (short)0, ao, false, false);
  }

  // D layout: M = v + (lane>=16)*8, N = lane&15. In-register LSTM update.
  const int mhi = (lane >> 4) * 8;
  const int j   = ntile * 16 + nb;             // hidden index
  #pragma unroll
  for (int v = 0; v < 8; ++v) {
    int mm = mtile * 16 + v + mhi;             // batch row
    float gi = sigmoid_f(ai[v] + bias[0 * Hn + j]);
    float gf = sigmoid_f(af[v] + bias[1 * Hn + j]);
    float gg = tanhf    (ag[v] + bias[2 * Hn + j]);
    float go = sigmoid_f(ao[v] + bias[3 * Hn + j]);
    size_t sidx = (size_t)mm * Hn + j;
    float cN = gf * c[sidx] + gi * gg;
    float hN = go * tanhf(cN);
    c[sidx] = cN;
    h[sidx] = hN;
    if (hcat) hcat[(size_t)mm * Kc + hoff + j] = (__bf16)hN;
  }
}

// ---------------------------------------------------------------------------
// Classifier: logits[32,32000] = Hcat[32,1024] @ Wcls^T + cls_b  (WMMA bf16)
// 4000 waves, one 16x16 output tile each; writes straight into d_out slice.
// ---------------------------------------------------------------------------
__global__ void __launch_bounds__(256) classifier_kernel(
    const __bf16* __restrict__ Hc, const __bf16* __restrict__ Wcls,
    const float* __restrict__ clsb, float* __restrict__ out, int step)
{
  const int wave  = blockIdx.x * 8 + (threadIdx.x >> 5);
  const int lane  = threadIdx.x & 31;
  const int mtile = wave & 1;
  const int ntile = wave >> 1;                  // 0..1999
  const int ncol  = ntile * 16;

  const int mrow  = mtile * 16 + (lane & 15);
  const int kbase = (lane >> 4) * 16;
  const int nb    = lane & 15;
  const int koffB = (lane >> 4) * 16;

  const __bf16* arow = Hc + (size_t)mrow * Kc;
  const __bf16* brow = Wcls + (size_t)(ncol + nb) * Kc;

  v8f acc = {};
  for (int k = 0; k < Kc; k += 32) {
    ABfrag a, b;
    a.h[0] = *(const v8bf*)(arow + k + kbase);
    a.h[1] = *(const v8bf*)(arow + k + kbase + 8);
    b.h[0] = *(const v8bf*)(brow + k + koffB);
    b.h[1] = *(const v8bf*)(brow + k + koffB + 8);
    acc = __builtin_amdgcn_wmma_f32_16x16x32_bf16(false, a.v, false, b.v, (short)0, acc, false, false);
  }

  const int mhi = (lane >> 4) * 8;
  const int col = ncol + nb;
  float*  orow  = out + (size_t)step * Vn;
  const float cb = clsb[col];
  #pragma unroll
  for (int v = 0; v < 8; ++v) {
    int mm = mtile * 16 + v + mhi;
    orow[(size_t)mm * ((size_t)Tn * Vn) + col] = acc[v] + cb;
  }
}

// ---------------------------------------------------------------------------
// Greedy argmax (first-max tie rule) over V=32000 logits per batch row.
// ---------------------------------------------------------------------------
__global__ void __launch_bounds__(256) argmax_kernel(
    const float* __restrict__ logits, int* __restrict__ out_tok)
{
  const int b = blockIdx.x;
  const float* row = logits + (size_t)b * ((size_t)Tn * Vn);
  float best = -INFINITY;
  int   bidx = Vn;
  for (int v = threadIdx.x; v < Vn; v += blockDim.x) {
    float x = row[v];
    if (x > best) { best = x; bidx = v; }      // ascending scan keeps first max
  }
  __shared__ float sv[256];
  __shared__ int   si[256];
  sv[threadIdx.x] = best; si[threadIdx.x] = bidx;
  __syncthreads();
  for (int s = 128; s > 0; s >>= 1) {
    if (threadIdx.x < (unsigned)s) {
      float ov = sv[threadIdx.x + s]; int oi = si[threadIdx.x + s];
      if (ov > sv[threadIdx.x] || (ov == sv[threadIdx.x] && oi < si[threadIdx.x])) {
        sv[threadIdx.x] = ov; si[threadIdx.x] = oi;
      }
    }
    __syncthreads();
  }
  if (threadIdx.x == 0) out_tok[b] = si[0];
}

// ---------------------------------------------------------------------------
extern "C" void kernel_launch(void* const* d_in, const int* in_sizes, int n_in,
                              void* d_out, int out_size, void* d_ws, size_t ws_size,
                              hipStream_t stream) {
  const int*   src      = (const int*)d_in[0];
  const int*   tgt      = (const int*)d_in[1];
  const float* enc_emb  = (const float*)d_in[2];
  const float* dec_emb  = (const float*)d_in[3];
  const float* e_wih_f  = (const float*)d_in[4];
  const float* e_whh_f  = (const float*)d_in[5];
  const float* e_bih_f  = (const float*)d_in[6];
  const float* e_bhh_f  = (const float*)d_in[7];
  const float* e_wih_r  = (const float*)d_in[8];
  const float* e_whh_r  = (const float*)d_in[9];
  const float* e_bih_r  = (const float*)d_in[10];
  const float* e_bhh_r  = (const float*)d_in[11];
  const float* d_wih_f  = (const float*)d_in[12];
  const float* d_whh_f  = (const float*)d_in[13];
  const float* d_bih_f  = (const float*)d_in[14];
  const float* d_bhh_f  = (const float*)d_in[15];
  const float* d_wih_r  = (const float*)d_in[16];
  const float* d_whh_r  = (const float*)d_in[17];
  const float* d_bih_r  = (const float*)d_in[18];
  const float* d_bhh_r  = (const float*)d_in[19];
  const float* cls_w    = (const float*)d_in[20];
  const float* cls_b    = (const float*)d_in[21];
  float* out = (float*)d_out;

  char* ws = (char*)d_ws;
  __bf16* Wc      = (__bf16*)(ws + OFF_WC);       // cells: 0=enc_f 1=enc_r 2=dec_f 3=dec_r
  __bf16* Wcls    = (__bf16*)(ws + OFF_WCLS);
  float*  bias    = (float*)(ws + OFF_BIAS);
  float*  hstate  = (float*)(ws + OFF_H);         // 2 cells
  float*  cstate  = (float*)(ws + OFF_C);         // 2 cells
  __bf16* Xcat    = (__bf16*)(ws + OFF_X);        // 2 buffers
  __bf16* Hcat    = (__bf16*)(ws + OFF_HCAT);
  int*    nexttok = (int*)(ws + OFF_TOK);

  __bf16* Wc_ef = Wc + 0 * WC_ELEMS;
  __bf16* Wc_er = Wc + 1 * WC_ELEMS;
  __bf16* Wc_df = Wc + 2 * WC_ELEMS;
  __bf16* Wc_dr = Wc + 3 * WC_ELEMS;
  float* bias_ef = bias + 0 * Gc;
  float* bias_er = bias + 1 * Gc;
  float* bias_df = bias + 2 * Gc;
  float* bias_dr = bias + 3 * Gc;
  float* h_f = hstate;              float* h_r = hstate + (size_t)Bn * Hn;
  float* c_f = cstate;              float* c_r = cstate + (size_t)Bn * Hn;
  __bf16* X0 = Xcat;                __bf16* X1 = Xcat + (size_t)Bn * Kc;

  // 1) Pack weights/biases to bf16, zero states + t=0 output slice.
  prep_kernel<<<4096, 256, 0, stream>>>(
      e_wih_f, e_whh_f, e_wih_r, e_whh_r, d_wih_f, d_whh_f, d_wih_r, d_whh_r,
      e_bih_f, e_bhh_f, e_bih_r, e_bhh_r, d_bih_f, d_bhh_f, d_bih_r, d_bhh_r,
      cls_w, Wc, Wcls, bias, hstate, out);

  // 2) Bidirectional encoder: 128 sequential steps, both directions per launch.
  for (int t = 0; t < Sn; ++t) {
    build_xcat2_kernel<<<dim3(Bn * Kc / 256, 2), 256, 0, stream>>>(
        enc_emb,
        src, Sn, t,            h_f, X0,
        src, Sn, Sn - 1 - t,   h_r, X1);
    lstm_cell2_kernel<<<dim3(8, 2), 256, 0, stream>>>(
        X0, Wc_ef, bias_ef, h_f, c_f, (__bf16*)nullptr, 0,
        X1, Wc_er, bias_er, h_r, c_r, (__bf16*)nullptr, 0);
  }

  // 3) Greedy decoder: 63 sequential steps (cells -> classifier -> argmax).
  for (int s = 1; s < Tn; ++s) {
    const int* tokp = (s == 1) ? tgt : nexttok;
    const int  tstr = (s == 1) ? Tn  : 1;
    build_xcat2_kernel<<<dim3(Bn * Kc / 256, 2), 256, 0, stream>>>(
        dec_emb,
        tokp, tstr, 0, h_f, X0,
        tokp, tstr, 0, h_r, X1);
    lstm_cell2_kernel<<<dim3(8, 2), 256, 0, stream>>>(
        X0, Wc_df, bias_df, h_f, c_f, Hcat, 0,
        X1, Wc_dr, bias_dr, h_r, c_r, Hcat, Hn);
    classifier_kernel<<<500, 256, 0, stream>>>(Hcat, Wcls, cls_b, out, s);
    argmax_kernel<<<Bn, 256, 0, stream>>>(out + (size_t)s * Vn, nexttok);
  }
}
